// MoEFFN_74904229642379
// MI455X (gfx1250) — matrix operand order repaired and checked by reading, not compiled, not run on hardware
//
#include <hip/hip_runtime.h>
#include <hip/hip_bf16.h>
#include <hip/amd_detail/amd_hip_unsafe_atomics.h>

// ---------------------------------------------------------------------------
// MoE FFN (top-2 of 8 experts), bf16 WMMA implementation for gfx1250.
//   B=4, S=2048 -> T=8192 tokens, D=1024, F=2048, E=8, K=2.
// Effective math = 16384 token-expert pairs * 3 GEMMs = ~206 GFLOP.
// bf16 weights (96MB) fit in the 192MB L2; fused SwiGLU keeps the 32x2048
// hidden tile in LDS (320KB/WGP) so H never touches HBM.
// ---------------------------------------------------------------------------

#define T_TOK   8192
#define DDIM    1024
#define FDIM    2048
#define NEXP    8
#define TILE_T  32            // tokens per FFN workgroup
#define MAXGRP  544           // >= sum ceil(n_e/32) worst case (<=519)

// ---- workspace layout (bytes) ----
#define OFF_WGT   0ull                         // [E][F][D] bf16  32MB
#define OFF_WUT   33554432ull                  // [E][F][D] bf16  32MB
#define OFF_WDT   67108864ull                  // [E][D][F] bf16  32MB
#define OFF_PROBS 100663296ull                 // [T][8] f32
#define OFF_LSE2  100925440ull                 // [T] f32
#define OFF_TOPI  100958208ull                 // [T][2] int
#define OFF_TOPW  101023744ull                 // [T][2] f32
#define OFF_TOKID 101089280ull                 // [E][T] int
#define OFF_TOKW  101351424ull                 // [E][T] f32
#define OFF_ECNT  101613568ull                 // [8] int
#define OFF_PSUM  101613632ull                 // [9] f32 (8 prob sums + lse2 sum)
#define OFF_NGRP  101613696ull                 // [1] int
#define OFF_GRPS  101613760ull                 // [MAXGRP] int

typedef __attribute__((ext_vector_type(16))) __bf16 bf16x16;
typedef __attribute__((ext_vector_type(8)))  float  floatx8;

__device__ __forceinline__ unsigned short f2bf(float f) {     // RNE
    unsigned int u = __float_as_uint(f);
    u += 0x7FFFu + ((u >> 16) & 1u);
    return (unsigned short)(u >> 16);
}

__device__ __forceinline__ floatx8 wmma_bf16(bf16x16 a, bf16x16 b, floatx8 c) {
    return __builtin_amdgcn_wmma_f32_16x16x32_bf16(false, a, false, b,
                                                   (short)0, c, false, false);
}

// A-fragment: two 16B chunks (lane-half selects contiguous 16-K half)
__device__ __forceinline__ bf16x16 load_frag2(const unsigned short* p0,
                                              const unsigned short* p1) {
    bf16x16 r;
    ((uint4*)&r)[0] = *(const uint4*)p0;
    ((uint4*)&r)[1] = *(const uint4*)p1;
    return r;
}
// B-fragment: 32 contiguous bytes (16 contiguous K of one column row in W^T)
__device__ __forceinline__ bf16x16 load_frag32B(const unsigned short* p) {
    bf16x16 r;
    ((uint4*)&r)[0] = ((const uint4*)p)[0];
    ((uint4*)&r)[1] = ((const uint4*)p)[1];
    return r;
}

// ---------------------------------------------------------------------------
__global__ void zero_kernel(float* out, int* ecnt) {
    size_t i = (size_t)blockIdx.x * blockDim.x + threadIdx.x;
    float4 z = {0.f, 0.f, 0.f, 0.f};
    ((float4*)out)[i] = z;                        // 8192*256 float4 = 8M floats
    if (blockIdx.x == 0 && threadIdx.x < NEXP) ecnt[threadIdx.x] = 0;
}

// src: [batch][R][C] f32   ->   dst: [batch][C][R] bf16
__global__ void transpose_cvt(const float* __restrict__ src,
                              unsigned short* __restrict__ dst, int R, int C) {
    __shared__ float tile[32][33];
    const float* s = src + (size_t)blockIdx.z * R * C;
    unsigned short* d = dst + (size_t)blockIdx.z * R * C;
    int c0 = blockIdx.x * 32, r0 = blockIdx.y * 32;
    for (int i = threadIdx.y; i < 32; i += 8)
        tile[i][threadIdx.x] = s[(size_t)(r0 + i) * C + c0 + threadIdx.x];
    __syncthreads();
    for (int i = threadIdx.y; i < 32; i += 8)
        d[(size_t)(c0 + i) * R + r0 + threadIdx.x] = f2bf(tile[threadIdx.x][i]);
}

// one wave (32 lanes) per token
__global__ void router_kernel(const float* __restrict__ x,
                              const float* __restrict__ Wr,
                              float* __restrict__ probs, float* __restrict__ lse2,
                              int* __restrict__ topi, float* __restrict__ topw) {
    int wave = threadIdx.x >> 5, lane = threadIdx.x & 31;
    int t = blockIdx.x * 8 + wave;
    float acc[NEXP] = {0.f, 0.f, 0.f, 0.f, 0.f, 0.f, 0.f, 0.f};
    const float* xr = x + (size_t)t * DDIM;
    for (int d = lane; d < DDIM; d += 32) {
        float xv = xr[d];
        const float* wr = Wr + d * NEXP;
        #pragma unroll
        for (int e = 0; e < NEXP; ++e) acc[e] += xv * wr[e];
    }
    #pragma unroll
    for (int e = 0; e < NEXP; ++e)
        for (int off = 16; off > 0; off >>= 1)
            acc[e] += __shfl_xor(acc[e], off, 32);
    if (lane == 0) {
        float m = acc[0];
        #pragma unroll
        for (int e = 1; e < NEXP; ++e) m = fmaxf(m, acc[e]);
        float p[NEXP], s = 0.f;
        #pragma unroll
        for (int e = 0; e < NEXP; ++e) { p[e] = __expf(acc[e] - m); s += p[e]; }
        float inv = 1.f / s;
        #pragma unroll
        for (int e = 0; e < NEXP; ++e) { p[e] *= inv; probs[t * NEXP + e] = p[e]; }
        float lse = m + __logf(s);
        lse2[t] = lse * lse;
        int i0 = 0; float b0 = p[0];
        #pragma unroll
        for (int e = 1; e < NEXP; ++e) if (p[e] > b0) { b0 = p[e]; i0 = e; }
        int i1 = -1; float b1 = -1.f;
        #pragma unroll
        for (int e = 0; e < NEXP; ++e)
            if (e != i0 && p[e] > b1) { b1 = p[e]; i1 = e; }
        float rs = 1.f / (b0 + b1);
        topi[t * 2] = i0; topi[t * 2 + 1] = i1;
        topw[t * 2] = b0 * rs; topw[t * 2 + 1] = b1 * rs;
    }
}

// single block, fixed-order (deterministic) reduction for aux-loss stats
__global__ void reduce_kernel(const float* __restrict__ probs,
                              const float* __restrict__ lse2,
                              float* __restrict__ psum) {
    __shared__ float s[256 * 9];
    int tid = threadIdx.x;
    float a[9] = {0.f, 0.f, 0.f, 0.f, 0.f, 0.f, 0.f, 0.f, 0.f};
    for (int t = tid; t < T_TOK; t += 256) {
        #pragma unroll
        for (int e = 0; e < NEXP; ++e) a[e] += probs[t * NEXP + e];
        a[8] += lse2[t];
    }
    #pragma unroll
    for (int j = 0; j < 9; ++j) s[tid * 9 + j] = a[j];
    __syncthreads();
    for (int off = 128; off > 0; off >>= 1) {
        if (tid < off)
            #pragma unroll
            for (int j = 0; j < 9; ++j) s[tid * 9 + j] += s[(tid + off) * 9 + j];
        __syncthreads();
    }
    if (tid == 0)
        #pragma unroll
        for (int j = 0; j < 9; ++j) psum[j] = s[j];
}

__global__ void dispatch_kernel(const int* __restrict__ topi,
                                const float* __restrict__ topw,
                                int* __restrict__ ecnt,
                                int* __restrict__ tokid, float* __restrict__ tokw) {
    int t = blockIdx.x * blockDim.x + threadIdx.x;
    #pragma unroll
    for (int k = 0; k < 2; ++k) {
        int e = topi[t * 2 + k];
        int slot = atomicAdd(&ecnt[e], 1);
        tokid[e * T_TOK + slot] = t;
        tokw[e * T_TOK + slot] = topw[t * 2 + k];
    }
}

// build (expert,group) table + finalize aux loss (single thread: deterministic)
__global__ void gb_kernel(const int* __restrict__ ecnt, int* __restrict__ groups,
                          int* __restrict__ ngroups, const float* __restrict__ psum,
                          float* __restrict__ out_aux) {
    if (threadIdx.x != 0 || blockIdx.x != 0) return;
    int total = 0;
    for (int e = 0; e < NEXP; ++e) {
        int ng = (ecnt[e] + TILE_T - 1) / TILE_T;
        for (int g = 0; g < ng; ++g) groups[total++] = (e << 16) | g;
    }
    *ngroups = total;
    float lb = 0.f;
    for (int e = 0; e < NEXP; ++e)
        lb += ((float)ecnt[e] / (float)(T_TOK * 2)) * (psum[e] / (float)T_TOK);
    float z = psum[8] / (float)T_TOK;
    out_aux[0] = 0.01f * (float)NEXP * lb + 0.001f * z;
}

// ---------------------------------------------------------------------------
// Fused expert FFN: 32 routed tokens per WG.
//   LDS: Xtile 32x1024 bf16 (padded pitch), Htile 32x2048 bf16 (padded pitch)
//   phase1: G/U = X*Wg / X*Wu  (WMMA bf16), H = silu(G)*U  -> LDS
//   phase2: Y = H*Wd (WMMA bf16), out[token] += w * Y (2 commutative fp adds)
// ---------------------------------------------------------------------------
#define XP (DDIM + 8)           // LDS pitches (ushorts), +16B to spread banks
#define HP (FDIM + 8)
#define SMEM_X   0
#define SMEM_H   (TILE_T * XP * 2)                    // 66048
#define SMEM_TOK (SMEM_H + TILE_T * HP * 2)           // 66048+131584=197632
#define SMEM_W   (SMEM_TOK + 128)
#define SMEM_BYTES (SMEM_W + 128)                     // 197888

__global__ __launch_bounds__(256, 1) void ffn_kernel(
    const float* __restrict__ x, float* __restrict__ out,
    const unsigned short* __restrict__ wgt, const unsigned short* __restrict__ wut,
    const unsigned short* __restrict__ wdt,
    const int* __restrict__ tokid, const float* __restrict__ tokw,
    const int* __restrict__ ecnt, const int* __restrict__ groups,
    const int* __restrict__ ngroups) {
    if ((int)blockIdx.x >= *ngroups) return;
    int ginfo = groups[blockIdx.x];
    int e = ginfo >> 16, g = ginfo & 0xFFFF;
    int cnt = ecnt[e];

    extern __shared__ char smem[];
    unsigned short* ldsX = (unsigned short*)(smem + SMEM_X);
    unsigned short* ldsH = (unsigned short*)(smem + SMEM_H);
    int*   ldsTok = (int*)(smem + SMEM_TOK);
    float* ldsW   = (float*)(smem + SMEM_W);

    int tid = threadIdx.x;
    if (tid < TILE_T) {
        int slot = g * TILE_T + tid;
        if (slot < cnt) {
            ldsTok[tid] = tokid[e * T_TOK + slot];
            ldsW[tid]   = tokw[e * T_TOK + slot];
        } else { ldsTok[tid] = -1; ldsW[tid] = 0.f; }
    }
    __syncthreads();

    // stage X tile (fp32 -> bf16)
    for (int i = tid; i < TILE_T * DDIM; i += 256) {
        int row = i >> 10, col = i & (DDIM - 1);
        int t = ldsTok[row];
        float v = (t >= 0) ? x[(size_t)t * DDIM + col] : 0.f;
        ldsX[row * XP + col] = f2bf(v);
    }
    __syncthreads();

    int wave = tid >> 5, lane = tid & 31;
    int lh = lane >> 4;          // lane half selects K sub-chunks
    int lm = lane & 15;

    // ---- phase 1: H = silu(X*Wg) * (X*Wu), wave owns 256 F-columns ----
    size_t wbase = (size_t)e * FDIM * DDIM;            // W^T [E][F][D]
    for (int c = 0; c < 8; ++c) {
        int n0 = wave * 256 + c * 32;
        floatx8 accg[2][2], accu[2][2];
        #pragma unroll
        for (int m = 0; m < 2; ++m)
            #pragma unroll
            for (int n = 0; n < 2; ++n) { accg[m][n] = (floatx8)0.f; accu[m][n] = (floatx8)0.f; }
        for (int k0 = 0; k0 < DDIM; k0 += 32) {
            bf16x16 A[2];
            #pragma unroll
            for (int m = 0; m < 2; ++m) {
                const unsigned short* ap = ldsX + (m * 16 + lm) * XP + k0 + lh * 16;
                A[m] = load_frag2(ap, ap + 8);
            }
            #pragma unroll
            for (int n = 0; n < 2; ++n) {
                size_t bo = wbase + (size_t)(n0 + n * 16 + lm) * DDIM + k0 + lh * 16;
                bf16x16 Bg = load_frag32B(wgt + bo);
                bf16x16 Bu = load_frag32B(wut + bo);
                #pragma unroll
                for (int m = 0; m < 2; ++m) {
                    accg[m][n] = wmma_bf16(A[m], Bg, accg[m][n]);
                    accu[m][n] = wmma_bf16(A[m], Bu, accu[m][n]);
                }
            }
        }
        #pragma unroll
        for (int m = 0; m < 2; ++m)
            #pragma unroll
            for (int n = 0; n < 2; ++n)
                #pragma unroll
                for (int r = 0; r < 8; ++r) {
                    float gv = accg[m][n][r], uv = accu[m][n][r];
                    float h = gv / (1.f + __expf(-gv)) * uv;   // SwiGLU
                    int row = m * 16 + r + lh * 8;
                    int col = n0 + n * 16 + lm;
                    ldsH[row * HP + col] = f2bf(h);
                }
    }
    __syncthreads();

    // ---- phase 2: Y = H*Wd, wave owns 128 D-columns ----
    size_t dbase = (size_t)e * DDIM * FDIM;            // Wd^T [E][D][F]
    floatx8 acc[2][8];
    #pragma unroll
    for (int m = 0; m < 2; ++m)
        #pragma unroll
        for (int n = 0; n < 8; ++n) acc[m][n] = (floatx8)0.f;
    for (int k0 = 0; k0 < FDIM; k0 += 32) {
        bf16x16 A[2];
        #pragma unroll
        for (int m = 0; m < 2; ++m) {
            const unsigned short* ap = ldsH + (m * 16 + lm) * HP + k0 + lh * 16;
            A[m] = load_frag2(ap, ap + 8);
        }
        #pragma unroll
        for (int n = 0; n < 8; ++n) {
            size_t bo = dbase + (size_t)(wave * 128 + n * 16 + lm) * FDIM + k0 + lh * 16;
            bf16x16 B = load_frag32B(wdt + bo);
            acc[0][n] = wmma_bf16(A[0], B, acc[0][n]);
            acc[1][n] = wmma_bf16(A[1], B, acc[1][n]);
        }
    }
    // weighted scatter: exactly 2 commutative fp32 atomic adds per out element
    int   trow[2][8];
    float twt[2][8];
    #pragma unroll
    for (int m = 0; m < 2; ++m)
        #pragma unroll
        for (int r = 0; r < 8; ++r) {
            int row = m * 16 + r + lh * 8;
            trow[m][r] = ldsTok[row];
            twt[m][r]  = ldsW[row];
        }
    #pragma unroll
    for (int m = 0; m < 2; ++m)
        #pragma unroll
        for (int n = 0; n < 8; ++n)
            #pragma unroll
            for (int r = 0; r < 8; ++r) {
                int t = trow[m][r];
                if (t >= 0)
                    unsafeAtomicAdd(&out[(size_t)t * DDIM + wave * 128 + n * 16 + lm],
                                    twt[m][r] * acc[m][n][r]);
            }
}

// ---------------------------------------------------------------------------
extern "C" void kernel_launch(void* const* d_in, const int* in_sizes, int n_in,
                              void* d_out, int out_size, void* d_ws, size_t ws_size,
                              hipStream_t stream) {
    const float* x  = (const float*)d_in[0];
    const float* Wr = (const float*)d_in[1];
    const float* Wg = (const float*)d_in[2];
    const float* Wu = (const float*)d_in[3];
    const float* Wd = (const float*)d_in[4];
    float* out = (float*)d_out;
    char* ws = (char*)d_ws;

    unsigned short* WGT = (unsigned short*)(ws + OFF_WGT);
    unsigned short* WUT = (unsigned short*)(ws + OFF_WUT);
    unsigned short* WDT = (unsigned short*)(ws + OFF_WDT);
    float* PROBS = (float*)(ws + OFF_PROBS);
    float* LSE2  = (float*)(ws + OFF_LSE2);
    int*   TOPI  = (int*)(ws + OFF_TOPI);
    float* TOPW  = (float*)(ws + OFF_TOPW);
    int*   TOKID = (int*)(ws + OFF_TOKID);
    float* TOKW  = (float*)(ws + OFF_TOKW);
    int*   ECNT  = (int*)(ws + OFF_ECNT);
    float* PSUM  = (float*)(ws + OFF_PSUM);
    int*   NGRP  = (int*)(ws + OFF_NGRP);
    int*   GRPS  = (int*)(ws + OFF_GRPS);

    zero_kernel<<<8192, 256, 0, stream>>>(out, ECNT);

    transpose_cvt<<<dim3(64, 32, 8), dim3(32, 8), 0, stream>>>(Wg, WGT, DDIM, FDIM);
    transpose_cvt<<<dim3(64, 32, 8), dim3(32, 8), 0, stream>>>(Wu, WUT, DDIM, FDIM);
    transpose_cvt<<<dim3(32, 64, 8), dim3(32, 8), 0, stream>>>(Wd, WDT, FDIM, DDIM);

    router_kernel<<<T_TOK / 8, 256, 0, stream>>>(x, Wr, PROBS, LSE2, TOPI, TOPW);
    reduce_kernel<<<1, 256, 0, stream>>>(PROBS, LSE2, PSUM);
    dispatch_kernel<<<T_TOK / 256, 256, 0, stream>>>(TOPI, TOPW, ECNT, TOKID, TOKW);
    gb_kernel<<<1, 64, 0, stream>>>(ECNT, GRPS, NGRP, PSUM, out + (size_t)T_TOK * DDIM);

    ffn_kernel<<<MAXGRP, 256, SMEM_BYTES, stream>>>(x, out, WGT, WUT, WDT,
                                                    TOKID, TOKW, ECNT, GRPS, NGRP);
}
